// GatedLightningIndexer_40535901340236
// MI455X (gfx1250) — matrix operand order repaired and checked by reading, not compiled, not run on hardware
//
#include <hip/hip_runtime.h>
#include <math.h>

#define HIDDEN  2048
#define DIDX    64
#define NHEADS  4
#define TSEQ    4096
#define QKC     320      // 256 q cols + 64 k cols, bf16
#define SCALE_F 0.125f   // 1/sqrt(64)

typedef __attribute__((ext_vector_type(16))) __bf16 v16bf;
typedef __attribute__((ext_vector_type(8)))  __bf16 v8bf;
typedef __attribute__((ext_vector_type(8)))  float  v8f;
typedef __attribute__((ext_vector_type(4)))  float  v4f;
typedef __attribute__((ext_vector_type(4)))  int    v4i;

#if defined(__HIP_DEVICE_COMPILE__) && \
    __has_builtin(__builtin_amdgcn_global_load_async_to_lds_b128) && \
    __has_builtin(__builtin_amdgcn_s_wait_asynccnt)
#define HAVE_ASYNC_LDS 1
typedef __attribute__((address_space(1))) v4i GV4;   // global v4i
typedef __attribute__((address_space(3))) v4i LV4;   // LDS v4i
#else
#define HAVE_ASYNC_LDS 0
#endif

#define CAT16 0,1,2,3,4,5,6,7,8,9,10,11,12,13,14,15

static __device__ __forceinline__ __bf16 f2bf(float f) { return (__bf16)f; }

static __device__ __forceinline__ float sigmoid_f(float x) {
  return 1.0f / (1.0f + __expf(-x));
}

// ---------------------------------------------------------------------------
// Kernel 0: f32 -> bf16 bulk convert (8 elements/thread, b128 traffic)
// ---------------------------------------------------------------------------
__global__ __launch_bounds__(256) void cvt_bf16_kernel(
    const float* __restrict__ src, __bf16* __restrict__ dst, long n) {
  long i = ((long)blockIdx.x * 256 + threadIdx.x) * 8;
  if (i >= n) return;
  v4f a = *(const v4f*)(src + i);
  v4f b = *(const v4f*)(src + i + 4);
  v8bf o;
  o[0]=f2bf(a.x); o[1]=f2bf(a.y); o[2]=f2bf(a.z); o[3]=f2bf(a.w);
  o[4]=f2bf(b.x); o[5]=f2bf(b.y); o[6]=f2bf(b.z); o[7]=f2bf(b.w);
  *(v8bf*)(dst + i) = o;
}

// ---------------------------------------------------------------------------
// Kernel 1: fused q/k projection, bf16 WMMA, all-bf16 inputs.
// One wave per 16-row x 64-col strip: 1 A-fragment + 4 independent
// B-fragments per K-chunk -> clause-able loads, 4 WMMAs, no cvts in loop.
// ---------------------------------------------------------------------------
__global__ __launch_bounds__(32) void proj_qk_kernel(
    const __bf16* __restrict__ xb, const __bf16* __restrict__ wqk,
    __bf16* __restrict__ qkt) {
  const int g    = blockIdx.x;        // 0..4  (64-col group)
  const int mt   = blockIdx.y;        // row tile
  const int lane = threadIdx.x;
  const int half = lane >> 4;
  const int ml   = lane & 15;

  const __bf16* Xrow = xb + (size_t)(mt * 16 + ml) * HIDDEN;
  const __bf16* Wr0 = wqk + (size_t)(g * 64 + 0 * 16 + ml) * HIDDEN;
  const __bf16* Wr1 = wqk + (size_t)(g * 64 + 1 * 16 + ml) * HIDDEN;
  const __bf16* Wr2 = wqk + (size_t)(g * 64 + 2 * 16 + ml) * HIDDEN;
  const __bf16* Wr3 = wqk + (size_t)(g * 64 + 3 * 16 + ml) * HIDDEN;

  v8f acc0 = {}, acc1 = {}, acc2 = {}, acc3 = {};
  #pragma unroll 2
  for (int k0 = 0; k0 < HIDDEN; k0 += 32) {
    // A fragment: e -> k = 16*half + e
    v8bf alo = *(const v8bf*)(Xrow + k0 + half * 16);
    v8bf ahi = *(const v8bf*)(Xrow + k0 + half * 16 + 8);
    // B fragments: e -> k = 16*half + e (contiguous 32B per lane)
    v16bf b0 = *(const v16bf*)(Wr0 + k0 + half * 16);
    v16bf b1 = *(const v16bf*)(Wr1 + k0 + half * 16);
    v16bf b2 = *(const v16bf*)(Wr2 + k0 + half * 16);
    v16bf b3 = *(const v16bf*)(Wr3 + k0 + half * 16);
    v16bf a = __builtin_shufflevector(alo, ahi, CAT16);
    acc0 = __builtin_amdgcn_wmma_f32_16x16x32_bf16(false, a, false, b0, (short)0, acc0, false, false);
    acc1 = __builtin_amdgcn_wmma_f32_16x16x32_bf16(false, a, false, b1, (short)0, acc1, false, false);
    acc2 = __builtin_amdgcn_wmma_f32_16x16x32_bf16(false, a, false, b2, (short)0, acc2, false, false);
    acc3 = __builtin_amdgcn_wmma_f32_16x16x32_bf16(false, a, false, b3, (short)0, acc3, false, false);
  }
  // C layout: lane -> n = lane&15 ; vgpr j + 8*half -> m
  __bf16* obase = qkt + (size_t)(mt * 16 + half * 8) * QKC + g * 64 + ml;
  #pragma unroll
  for (int j = 0; j < 8; ++j) {
    __bf16* orow = obase + (size_t)j * QKC;
    orow[0]  = f2bf(acc0[j]);
    orow[16] = f2bf(acc1[j]);
    orow[32] = f2bf(acc2[j]);
    orow[48] = f2bf(acc3[j]);
  }
}

// ---------------------------------------------------------------------------
// Kernel 2: per-token head gate wsig[tok][h] = sigmoid(x.Ww[h] + bw[h])
// ---------------------------------------------------------------------------
__global__ __launch_bounds__(256) void proj_w_kernel(
    const float* __restrict__ x, const float* __restrict__ Ww,
    const float* __restrict__ bw, float* __restrict__ wsig, int M) {
  const int wave = threadIdx.x >> 5;
  const int lane = threadIdx.x & 31;
  const int tok  = blockIdx.x * 8 + wave;
  if (tok >= M) return;
  const float* Xr = x + (size_t)tok * HIDDEN;
  float p0 = 0.f, p1 = 0.f, p2 = 0.f, p3 = 0.f;
  for (int i = lane; i < HIDDEN; i += 32) {
    float xv = Xr[i];
    p0 += xv * Ww[0 * HIDDEN + i];
    p1 += xv * Ww[1 * HIDDEN + i];
    p2 += xv * Ww[2 * HIDDEN + i];
    p3 += xv * Ww[3 * HIDDEN + i];
  }
  #pragma unroll
  for (int s = 16; s > 0; s >>= 1) {
    p0 += __shfl_xor(p0, s, 32);
    p1 += __shfl_xor(p1, s, 32);
    p2 += __shfl_xor(p2, s, 32);
    p3 += __shfl_xor(p3, s, 32);
  }
  if (lane == 0) {
    wsig[(size_t)tok * 4 + 0] = sigmoid_f(p0 + bw[0]);
    wsig[(size_t)tok * 4 + 1] = sigmoid_f(p1 + bw[1]);
    wsig[(size_t)tok * 4 + 2] = sigmoid_f(p2 + bw[2]);
    wsig[(size_t)tok * 4 + 3] = sigmoid_f(p3 + bw[3]);
  }
}

// ---------------------------------------------------------------------------
// Kernel 3: scores. 128x128 tile per block (8 waves), WMMA bf16 q.k^T per
// head, sigmoid gate, head-sum, causal mask. Output via non-temporal stores.
// ---------------------------------------------------------------------------
#define KLSTRIDE 80   // 64 + 16 bf16 pad (160B rows, 32B-aligned vector reads)

__global__ __launch_bounds__(256) void scores_kernel(
    const __bf16* __restrict__ qkt, const float* __restrict__ wsig,
    const float* __restrict__ idx_bias, float* __restrict__ out) {
  const int ct = blockIdx.x, rt = blockIdx.y, b = blockIdx.z;
  const int T = TSEQ;
  float* outb = out + (size_t)b * T * T;
  const int tid = threadIdx.x;
  const int rowbase = rt * 128, colbase = ct * 128;

  if (ct > rt) {  // tile entirely above the diagonal -> all -inf, NT stores
    const v4f ninf4 = {-INFINITY, -INFINITY, -INFINITY, -INFINITY};
    #pragma unroll
    for (int i = 0; i < 16; ++i) {
      int idx = i * 256 + tid;          // 4096 float4's
      int r = idx >> 5, c = (idx & 31) * 4;
      __builtin_nontemporal_store(
          ninf4, (v4f*)&outb[(size_t)(rowbase + r) * T + colbase + c]);
    }
    return;
  }

  __shared__ __bf16 kls[128 * KLSTRIDE];
  const __bf16* qb = qkt + (size_t)b * T * QKC;

  // Stage K tile: 128 tokens x 64 dims bf16 -> LDS (async if available)
  {
    int tokl = tid >> 1, hh = tid & 1;  // each thread: 32 bf16 (64B)
    const __bf16* src = qb + (size_t)(colbase + tokl) * QKC + 256 + hh * 32;
    __bf16* dst = &kls[tokl * KLSTRIDE + hh * 32];
#if HAVE_ASYNC_LDS
    #pragma unroll
    for (int q = 0; q < 4; ++q)
      __builtin_amdgcn_global_load_async_to_lds_b128(
          (GV4*)(src + q * 8), (LV4*)(dst + q * 8), 0, 0);
    __builtin_amdgcn_s_wait_asynccnt(0);
#else
    const v8bf* s8 = (const v8bf*)src;
    v8bf* d8 = (v8bf*)dst;
    d8[0] = s8[0]; d8[1] = s8[1]; d8[2] = s8[2]; d8[3] = s8[3];
#endif
  }
  __syncthreads();

  const int wv = tid >> 5, lane = tid & 31;
  const int half = lane >> 4, ln = lane & 15;
  const int rowb = rowbase + wv * 16;

  // Q fragments for this wave's 16 rows: 4 heads x 2 K-chunks, kept in VGPRs
  v16bf aq[NHEADS][2];
  {
    const __bf16* qrow = qb + (size_t)(rowb + ln) * QKC;
    #pragma unroll
    for (int h = 0; h < NHEADS; ++h)
      #pragma unroll
      for (int c = 0; c < 2; ++c) {
        v8bf lo = *(const v8bf*)(qrow + h * 64 + c * 32 + half * 16);
        v8bf hi = *(const v8bf*)(qrow + h * 64 + c * 32 + half * 16 + 8);
        aq[h][c] = __builtin_shufflevector(lo, hi, CAT16);
      }
  }

  // Head gates for this lane's 8 output rows (m = j + 8*half)
  float wg[8][NHEADS];
  #pragma unroll
  for (int j = 0; j < 8; ++j) {
    float4 w4 = *(const float4*)(wsig + (size_t)(b * T + rowb + j + half * 8) * 4);
    wg[j][0] = w4.x; wg[j][1] = w4.y; wg[j][2] = w4.z; wg[j][3] = w4.w;
  }
  const float ib[NHEADS] = {idx_bias[0], idx_bias[1], idx_bias[2], idx_bias[3]};
  const bool diag = (ct == rt);

  #pragma unroll 1
  for (int sub = 0; sub < 8; ++sub) {
    const int colb = colbase + sub * 16;
    // B fragments shared across all 4 heads: k = 16*half + e, 32B aligned
    const __bf16* krow = &kls[(sub * 16 + ln) * KLSTRIDE];
    v16bf bfrag0 = *(const v16bf*)(krow + half * 16);
    v16bf bfrag1 = *(const v16bf*)(krow + 32 + half * 16);

    float fin[8] = {0.f, 0.f, 0.f, 0.f, 0.f, 0.f, 0.f, 0.f};
    #pragma unroll
    for (int h = 0; h < NHEADS; ++h) {
      v8f acc = {};
      acc = __builtin_amdgcn_wmma_f32_16x16x32_bf16(false, aq[h][0], false, bfrag0,
                                                    (short)0, acc, false, false);
      acc = __builtin_amdgcn_wmma_f32_16x16x32_bf16(false, aq[h][1], false, bfrag1,
                                                    (short)0, acc, false, false);
      #pragma unroll
      for (int j = 0; j < 8; ++j)
        fin[j] += wg[j][h] * sigmoid_f(acc[j] * SCALE_F + ib[h]);
    }
    #pragma unroll
    for (int j = 0; j < 8; ++j) {
      int row = rowb + j + half * 8;
      int col = colb + ln;
      float v = (diag && col > row) ? -INFINITY : fin[j];
      __builtin_nontemporal_store(v, &outb[(size_t)row * T + col]);
    }
  }
}

// ---------------------------------------------------------------------------
extern "C" void kernel_launch(void* const* d_in, const int* in_sizes, int n_in,
                              void* d_out, int out_size, void* d_ws, size_t ws_size,
                              hipStream_t stream) {
  const float* x  = (const float*)d_in[0];
  const float* Wq = (const float*)d_in[1];
  const float* Wk = (const float*)d_in[2];
  const float* Ww = (const float*)d_in[3];
  const float* bw = (const float*)d_in[4];
  const float* ib = (const float*)d_in[5];

  const int M = in_sizes[0] / HIDDEN;   // B*T tokens
  const int B = M / TSEQ;

  // Workspace carve (all 16B aligned):
  char* p = (char*)d_ws;
  __bf16* xb  = (__bf16*)p;  p += (size_t)M * HIDDEN * sizeof(__bf16);   // 32MB
  __bf16* wqk = (__bf16*)p;  p += (size_t)QKC * HIDDEN * sizeof(__bf16); // 1.25MB
  __bf16* qkt = (__bf16*)p;  p += (size_t)M * QKC * sizeof(__bf16);      // 5.25MB
  float*  wsg = (float*)p;

  const long nx = (long)M * HIDDEN;
  cvt_bf16_kernel<<<dim3((unsigned)(nx / (256 * 8))), 256, 0, stream>>>(x, xb, nx);
  cvt_bf16_kernel<<<dim3(256 * HIDDEN / (256 * 8)), 256, 0, stream>>>(
      Wq, wqk, (long)256 * HIDDEN);
  cvt_bf16_kernel<<<dim3(64 * HIDDEN / (256 * 8)), 256, 0, stream>>>(
      Wk, wqk + (size_t)256 * HIDDEN, (long)64 * HIDDEN);

  proj_qk_kernel<<<dim3(QKC / 64, M / 16), 32, 0, stream>>>(xb, wqk, qkt);
  proj_w_kernel<<<dim3((M + 7) / 8), 256, 0, stream>>>(x, Ww, bw, wsg, M);
  scores_kernel<<<dim3(TSEQ / 128, TSEQ / 128, B), 256, 0, stream>>>(
      qkt, wsg, ib, (float*)d_out);
}
